// WanTransformerBlock_15659450761690
// MI455X (gfx1250) — matrix operand order repaired and checked by reading, not compiled, not run on hardware
//
#include <hip/hip_runtime.h>

// ---------------- problem constants ----------------
#define DIMC 1536
#define HN   12
#define DHC  128
#define FNUM 2
#define TPFC 880
#define SN   (FNUM * TPFC)   // 1760 tokens
#define SCC  TPFC            // 880 cached kv
#define SKVN (SCC + SN)      // 2640 total keys (self-attn)
#define FFNN 8960
#define LEN  512
#define EPSF 1e-6f
#define ATT_SCALE 0.08838834764831845f   // 1/sqrt(128)

typedef __attribute__((ext_vector_type(16))) _Float16 v16h;
typedef __attribute__((ext_vector_type(8)))  float    v8f;
typedef __attribute__((ext_vector_type(4)))  unsigned int u32x4;
typedef __attribute__((ext_vector_type(4)))  int      i32x4;
typedef __attribute__((ext_vector_type(8)))  int      i32x8;

// Tensor Data Mover availability (device pass only; arity differs by toolchain)
#if defined(__HIP_DEVICE_COMPILE__) && __has_builtin(__builtin_amdgcn_tensor_load_to_lds)
#define TDM_OK 1
#if __has_include(<hip/amd_detail/amd_gfx1250_TDM.h>)
#define TDM_6ARG 1
#endif
#endif

// ---------------- WMMA helpers ----------------
__device__ inline v8f wmma_f16(v16h a, v16h b, v8f c) {
  // D = A(16x32 f16) * B(32x16 f16) + C(16x16 f32)
  return __builtin_amdgcn_wmma_f32_16x16x32_f16(
      false, a, false, b, (short)0, c, false, false);
}

// A-fragment (16x32 f16) element mapping used here (contiguous k, not the ISA table order):
// lane L holds row m = L&15 ; k = (L>>4)*16 + s (matches B fragment order).
__device__ inline int a_frag_k(int lane, int s) {
  int half = lane >> 4;
  return half * 16 + s;
}
// B-fragment (32x16 f16): lane L holds col n = L&15, k = (L>>4)*16 + s.
// C-matrix (16x16 f32):   lane L, reg r -> m = (L>>4)*8 + r, n = L&15.

__device__ inline float gelu_tanh(float x) {
  float x3 = x * x * x;
  return 0.5f * x * (1.f + tanhf(0.7978845608f * (x + 0.044715f * x3)));
}

#if defined(TDM_OK)
// Issue a 2D TDM tile load: global (row-major, stride0 elems/row) -> LDS,
// zero-fill beyond (tensor_d0 x tensor_d1). data_size_log: 0=1B,1=2B,2=4B.
// D# packing per CDNA5 ISA 8.3/8.4 (group0 128b, group1 256b, groups 2/3 zero).
__device__ __forceinline__ void tdm_load_2d(
    const void* gaddr, unsigned lds_off, int data_size_log,
    int tile_d0, int tile_d1,
    unsigned long long tensor_d0, unsigned long long tensor_d1,
    unsigned long long stride0)
{
  unsigned long long ga = (unsigned long long)gaddr;
  u32x4 g0;
  g0[0] = 1u;                                    // count=1, user descriptor
  g0[1] = lds_off;                               // lds_addr (bytes)
  g0[2] = (unsigned)(ga & 0xFFFFFFFFu);          // global_addr[31:0]
  g0[3] = (unsigned)((ga >> 32) & 0x1FFFFFFu)    // global_addr[56:32]
        | (2u << 30);                            // type = 2 ("image")
  i32x8 g1;
  g1[0] = (int)((unsigned)data_size_log << 16);              // data_size
  g1[1] = (int)((unsigned)(tensor_d0 & 0xFFFFu) << 16);      // tensor_dim0[15:0]
  g1[2] = (int)(((tensor_d0 >> 16) & 0xFFFFu) |
                ((tensor_d1 & 0xFFFFu) << 16));              // dim0 hi | dim1 lo
  g1[3] = (int)(((tensor_d1 >> 16) & 0xFFFFu) |
                ((unsigned)tile_d0 << 16));                  // dim1 hi | tile0
  g1[4] = (int)((unsigned)tile_d1);                          // tile1 | tile2=0
  g1[5] = (int)(stride0 & 0xFFFFFFFFu);                      // dim0 stride lo
  g1[6] = (int)((stride0 >> 32) & 0xFFFFu);                  // stride hi | s1=0
  g1[7] = 0;
  i32x4 z4 = {0, 0, 0, 0};
#if defined(TDM_6ARG)
  i32x8 z8 = {0, 0, 0, 0, 0, 0, 0, 0};
  __builtin_amdgcn_tensor_load_to_lds(g0, g1, z4, z4, z8, 0);
#else
  __builtin_amdgcn_tensor_load_to_lds(g0, g1, z4, z4, 0);
#endif
}
__device__ __forceinline__ unsigned lds_offset_of(const void* p) {
  return (unsigned)(unsigned long long)(size_t)p;  // aperture low 32 bits = LDS byte offset
}
#endif

// ---------------- generic WMMA GEMM ----------------
// C[M,N] = epilogue( A_f16[M,K] @ W_f32[K,N] + bias )
// OUT_F16: store f16 ; EPI: 0=store f32, 1=gelu(store), 2=accumulate f32 (+=)
// TDM path: double-buffered tensor_load_to_lds of row-major tiles, then an
// LDS-only shuffle into WMMA fragment layout; fallback: global gathers.
template <bool OUT_F16, int EPI>
__global__ __launch_bounds__(256) void gemm_kernel(
    const _Float16* __restrict__ A, const float* __restrict__ W,
    const float* __restrict__ bias, void* __restrict__ outp,
    int M, int N, int K)
{
  __shared__ __align__(64) _Float16 la[8 * 512];  // 8 m-subtiles, frag layout
  __shared__ __align__(64) _Float16 lb[4 * 512];  // 4 n-subtiles, frag layout
  const int gm0 = blockIdx.x * 128, gn0 = blockIdx.y * 64;
  const int tid = threadIdx.x, lane = tid & 31, wave = tid >> 5;
  const int wm = wave >> 1, wn = wave & 1;   // 4x2 wave grid -> 32x32 per wave

  v8f acc[2][2] = {};

#if defined(TDM_OK)
  __shared__ __align__(64) _Float16 sA[2][128 * 32];  // 2x8KB  row-major A tiles
  __shared__ __align__(64) float    sW[2][32 * 64];   // 2x8KB  row-major W tiles
  // prologue: TDM-load tile k0=0 (wave 0 issues; TENSORcnt tracks)
  if (tid < 32) {
    tdm_load_2d(&A[(size_t)gm0 * K], lds_offset_of(&sA[0][0]), 1,
                /*tile*/32, 128, /*tensor*/(unsigned long long)K,
                (unsigned long long)(M - gm0), (unsigned long long)K);
    tdm_load_2d(&W[gn0], lds_offset_of(&sW[0][0]), 2,
                /*tile*/64, 32, /*tensor*/(unsigned long long)(N - gn0),
                (unsigned long long)K, (unsigned long long)N);
  }
  int cur = 0;
  for (int k0 = 0; k0 < K; k0 += 32) {
    if (tid < 32) __builtin_amdgcn_s_wait_tensorcnt(0);
    __syncthreads();          // scratch[cur] ready; frag regs of prev iter consumed
    if (k0 + 32 < K && tid < 32) {  // overlap next tile DMA with shuffle+compute
      tdm_load_2d(&A[(size_t)gm0 * K + (k0 + 32)],
                  lds_offset_of(&sA[cur ^ 1][0]), 1, 32, 128,
                  (unsigned long long)(K - (k0 + 32)),
                  (unsigned long long)(M - gm0), (unsigned long long)K);
      tdm_load_2d(&W[(size_t)(k0 + 32) * N + gn0],
                  lds_offset_of(&sW[cur ^ 1][0]), 2, 64, 32,
                  (unsigned long long)(N - gn0),
                  (unsigned long long)(K - (k0 + 32)), (unsigned long long)N);
    }
    // LDS shuffle: row-major scratch -> fragment layout (OOB rows TDM-zeroed)
    #pragma unroll 4
    for (int e = tid; e < 4096; e += 256) {
      int sub = e >> 9, r = e & 511, L = r & 31, s = r >> 5;
      la[sub * 512 + L * 16 + s] =
          sA[cur][(sub * 16 + (L & 15)) * 32 + a_frag_k(L, s)];
    }
    #pragma unroll 4
    for (int e = tid; e < 2048; e += 256) {
      int sub = e >> 9, r = e & 511, L = r & 31, s = r >> 5;
      lb[sub * 512 + L * 16 + s] =
          (_Float16)sW[cur][(((L >> 4) << 4) + s) * 64 + sub * 16 + (L & 15)];
    }
    __syncthreads();
    v16h a0 = *(const v16h*)&la[(wm * 2 + 0) * 512 + lane * 16];
    v16h a1 = *(const v16h*)&la[(wm * 2 + 1) * 512 + lane * 16];
    v16h b0 = *(const v16h*)&lb[(wn * 2 + 0) * 512 + lane * 16];
    v16h b1 = *(const v16h*)&lb[(wn * 2 + 1) * 512 + lane * 16];
    acc[0][0] = wmma_f16(a0, b0, acc[0][0]);
    acc[0][1] = wmma_f16(a0, b1, acc[0][1]);
    acc[1][0] = wmma_f16(a1, b0, acc[1][0]);
    acc[1][1] = wmma_f16(a1, b1, acc[1][1]);
    cur ^= 1;
  }
#else
  for (int k0 = 0; k0 < K; k0 += 32) {
    __syncthreads();
    #pragma unroll 4
    for (int e = tid; e < 4096; e += 256) {
      int sub = e >> 9, r = e & 511, L = r & 31, s = r >> 5;
      int m = gm0 + sub * 16 + (L & 15);
      int kk = k0 + a_frag_k(L, s);
      la[sub * 512 + L * 16 + s] =
          (m < M) ? A[(size_t)m * K + kk] : (_Float16)0.f;
    }
    #pragma unroll 4
    for (int e = tid; e < 2048; e += 256) {
      int sub = e >> 9, r = e & 511, L = r & 31, s = r >> 5;
      int n = gn0 + sub * 16 + (L & 15);
      int kk = k0 + ((L >> 4) << 4) + s;
      lb[sub * 512 + L * 16 + s] = (_Float16)W[(size_t)kk * N + n];
    }
    if (k0 + 32 < K) {  // warm L2 for the next K tile (global_prefetch_b8)
      __builtin_prefetch(&W[(size_t)(k0 + 32) * N + gn0 + (tid & 63)], 0, 1);
    }
    __syncthreads();
    v16h a0 = *(const v16h*)&la[(wm * 2 + 0) * 512 + lane * 16];
    v16h a1 = *(const v16h*)&la[(wm * 2 + 1) * 512 + lane * 16];
    v16h b0 = *(const v16h*)&lb[(wn * 2 + 0) * 512 + lane * 16];
    v16h b1 = *(const v16h*)&lb[(wn * 2 + 1) * 512 + lane * 16];
    acc[0][0] = wmma_f16(a0, b0, acc[0][0]);
    acc[0][1] = wmma_f16(a0, b1, acc[0][1]);
    acc[1][0] = wmma_f16(a1, b0, acc[1][0]);
    acc[1][1] = wmma_f16(a1, b1, acc[1][1]);
  }
#endif

  // epilogue (C layout: lane holds rows (lane>>4)*8+r, col lane&15)
  #pragma unroll
  for (int i = 0; i < 2; i++)
    #pragma unroll
    for (int j = 0; j < 2; j++) {
      int mbase = gm0 + (wm * 2 + i) * 16 + ((lane >> 4) * 8);
      int n = gn0 + (wn * 2 + j) * 16 + (lane & 15);
      float bv = bias ? bias[n] : 0.f;
      #pragma unroll
      for (int r = 0; r < 8; r++) {
        int m = mbase + r;
        if (m >= M) continue;
        float v = acc[i][j][r] + bv;
        if (EPI == 1) v = gelu_tanh(v);
        if (OUT_F16) {
          ((_Float16*)outp)[(size_t)m * N + n] = (_Float16)v;
        } else if (EPI == 2) {
          ((float*)outp)[(size_t)m * N + n] += v;
        } else {
          ((float*)outp)[(size_t)m * N + n] = v;
        }
      }
    }
}

// ---------------- flash attention (WMMA QK^T and PV) ----------------
// Q: [BH][Sq][128] f16, K/V: [BH][Skv][128] f16, O: [BH][Sq][128] f32
__global__ __launch_bounds__(128) void flash_kernel(
    const _Float16* __restrict__ Q, const _Float16* __restrict__ Km,
    const _Float16* __restrict__ Vm, float* __restrict__ O,
    int Sq, int Skv, float scale)
{
  const int bh = blockIdx.x;
  const int q0 = blockIdx.y * 64;
  const _Float16* Qb = Q + (size_t)bh * Sq * DHC;
  const _Float16* Kb = Km + (size_t)bh * Skv * DHC;
  const _Float16* Vb = Vm + (size_t)bh * Skv * DHC;
  float* Ob = O + (size_t)bh * Sq * DHC;
  const int tid = threadIdx.x, lane = tid & 31, wave = tid >> 5;

  __shared__ __align__(64) _Float16 Qs[64 * DHC];     // 16KB row-major q tile
  __shared__ __align__(64) _Float16 Kf[4][2][512];    // 8KB  K frags
  __shared__ __align__(64) _Float16 Vf[8][512];       // 8KB  V frags
  __shared__ __align__(64) _Float16 Ps[4][16 * 32];   // 4KB  wave-private P scratch

  for (int e = tid; e < 64 * 16; e += 128) {
    int row = e >> 4, c8 = (e & 15) * 8;
    int m = q0 + row;
    if (m < Sq) {
      *(uint4*)&Qs[row * DHC + c8] = *(const uint4*)&Qb[(size_t)m * DHC + c8];
    } else {
      uint4 z = {0u, 0u, 0u, 0u};
      *(uint4*)&Qs[row * DHC + c8] = z;
    }
  }
  __syncthreads();

  // per-wave Q A-fragments, resident for the whole key loop
  v16h qf[4];
  #pragma unroll
  for (int kk = 0; kk < 4; kk++) {
    v16h q;
    #pragma unroll
    for (int s = 0; s < 16; s++)
      q[s] = Qs[(wave * 16 + (lane & 15)) * DHC + kk * 32 + a_frag_k(lane, s)];
    qf[kk] = q;
  }

  float mrow[8], lrow[8];
  #pragma unroll
  for (int r = 0; r < 8; r++) { mrow[r] = -1e30f; lrow[r] = 0.f; }
  v8f oacc[8] = {};

  for (int c0 = 0; c0 < Skv; c0 += 32) {
    __syncthreads();
    for (int e = tid; e < 4096; e += 128) {
      int kk = e >> 10, r2 = e & 1023, sub = r2 >> 9, r3 = r2 & 511;
      int L = r3 & 31, s = r3 >> 5;
      int key = c0 + sub * 16 + (L & 15);
      int d = kk * 32 + ((L >> 4) << 4) + s;
      Kf[kk][sub][L * 16 + s] =
          (key < Skv) ? Kb[(size_t)key * DHC + d] : (_Float16)0.f;
    }
    for (int e = tid; e < 4096; e += 128) {
      int sub = e >> 9, r3 = e & 511, L = r3 & 31, s = r3 >> 5;
      int key = c0 + ((L >> 4) << 4) + s;
      int d = sub * 16 + (L & 15);
      Vf[sub][L * 16 + s] =
          (key < Skv) ? Vb[(size_t)key * DHC + d] : (_Float16)0.f;
    }
    __syncthreads();

    // preload ALL K fragments, then run the WMMA chain (waits amortized)
    v16h kb[8];
    #pragma unroll
    for (int kk = 0; kk < 4; kk++) {
      kb[kk * 2 + 0] = *(const v16h*)&Kf[kk][0][lane * 16];
      kb[kk * 2 + 1] = *(const v16h*)&Kf[kk][1][lane * 16];
    }
    v8f sacc[2] = {};
    #pragma unroll
    for (int kk = 0; kk < 4; kk++) {
      sacc[0] = wmma_f16(qf[kk], kb[kk * 2 + 0], sacc[0]);
      sacc[1] = wmma_f16(qf[kk], kb[kk * 2 + 1], sacc[1]);
    }

    // online softmax; rows live across 16-lane groups (C layout)
    int col0 = c0 + (lane & 15), col1 = col0 + 16;
    float mnew[8];
    #pragma unroll
    for (int r = 0; r < 8; r++) {
      float s0 = sacc[0][r] * scale, s1 = sacc[1][r] * scale;
      if (col0 >= Skv) s0 = -1e30f;
      if (col1 >= Skv) s1 = -1e30f;
      sacc[0][r] = s0; sacc[1][r] = s1;
      float mx = fmaxf(s0, s1);
      #pragma unroll
      for (int off = 1; off < 16; off <<= 1) mx = fmaxf(mx, __shfl_xor(mx, off, 16));
      mnew[r] = fmaxf(mrow[r], mx);
    }
    #pragma unroll
    for (int r = 0; r < 8; r++) {
      float corr = __expf(mrow[r] - mnew[r]);
      float p0 = __expf(sacc[0][r] - mnew[r]);
      float p1 = __expf(sacc[1][r] - mnew[r]);
      float ps = p0 + p1;
      #pragma unroll
      for (int off = 1; off < 16; off <<= 1) ps += __shfl_xor(ps, off, 16);
      lrow[r] = lrow[r] * corr + ps;
      mrow[r] = mnew[r];
      #pragma unroll
      for (int sub = 0; sub < 8; sub++) oacc[sub][r] *= corr;
      int row = (lane >> 4) * 8 + r;
      Ps[wave][row * 32 + (lane & 15)] = (_Float16)p0;
      Ps[wave][row * 32 + 16 + (lane & 15)] = (_Float16)p1;
    }

    // re-fragment P, preload V fragments, then the PV WMMA chain
    v16h pf;
    #pragma unroll
    for (int s = 0; s < 16; s++)
      pf[s] = Ps[wave][(lane & 15) * 32 + a_frag_k(lane, s)];
    v16h vb[8];
    #pragma unroll
    for (int sub = 0; sub < 8; sub++)
      vb[sub] = *(const v16h*)&Vf[sub][lane * 16];
    #pragma unroll
    for (int sub = 0; sub < 8; sub++)
      oacc[sub] = wmma_f16(pf, vb[sub], oacc[sub]);
  }

  #pragma unroll
  for (int sub = 0; sub < 8; sub++)
    #pragma unroll
    for (int r = 0; r < 8; r++) {
      int m = q0 + wave * 16 + (lane >> 4) * 8 + r;
      if (m < Sq)
        Ob[(size_t)m * DHC + sub * 16 + (lane & 15)] = oacc[sub][r] / lrow[r];
    }
}

// ---------------- LayerNorm (+optional affine, +optional AdaLN mod) ----------------
__global__ __launch_bounds__(256) void layernorm_kernel(
    const float* __restrict__ x, _Float16* __restrict__ y,
    const float* __restrict__ w, const float* __restrict__ b,
    const float* __restrict__ sstT, const float* __restrict__ temb,
    int scaleRow, int shiftRow)
{
  __shared__ float red[256];
  const int row = blockIdx.x, tid = threadIdx.x;
  const float* xr = x + (size_t)row * DIMC;
  float s = 0.f;
  for (int c = tid; c < DIMC; c += 256) s += xr[c];
  red[tid] = s; __syncthreads();
  for (int st = 128; st > 0; st >>= 1) { if (tid < st) red[tid] += red[tid + st]; __syncthreads(); }
  float mu = red[0] / DIMC; __syncthreads();
  float v = 0.f;
  for (int c = tid; c < DIMC; c += 256) { float d = xr[c] - mu; v += d * d; }
  red[tid] = v; __syncthreads();
  for (int st = 128; st > 0; st >>= 1) { if (tid < st) red[tid] += red[tid + st]; __syncthreads(); }
  float inv = rsqrtf(red[0] / DIMC + EPSF);
  for (int c = tid; c < DIMC; c += 256) {
    float val = (xr[c] - mu) * inv;
    if (w) val = val * w[c] + b[c];
    if (sstT) {
      float sc = sstT[scaleRow * DIMC + c] + temb[scaleRow * DIMC + c];
      float sh = sstT[shiftRow * DIMC + c] + temb[shiftRow * DIMC + c];
      val = val * (1.f + sc) + sh;
    }
    y[(size_t)row * DIMC + c] = (_Float16)val;
  }
}

// ---------------- full-width RMS norm (in-place, f32) ----------------
__global__ __launch_bounds__(256) void rms_kernel(float* __restrict__ x,
                                                  const float* __restrict__ w)
{
  __shared__ float red[256];
  const int row = blockIdx.x, tid = threadIdx.x;
  float* xr = x + (size_t)row * DIMC;
  float s = 0.f;
  for (int c = tid; c < DIMC; c += 256) { float v = xr[c]; s += v * v; }
  red[tid] = s; __syncthreads();
  for (int st = 128; st > 0; st >>= 1) { if (tid < st) red[tid] += red[tid + st]; __syncthreads(); }
  float inv = rsqrtf(red[0] / DIMC + EPSF);
  for (int c = tid; c < DIMC; c += 256) xr[c] = xr[c] * inv * w[c];
}

// ---------------- P / P^-1 (affine 4x4) ----------------
__global__ void prope_mats_kernel(const float* __restrict__ vm,
                                  const float* __restrict__ Ksm,
                                  float* __restrict__ P, float* __restrict__ Pinv)
{
  int f = threadIdx.x;
  if (f >= FNUM) return;
  float K4[16];
  for (int i = 0; i < 16; i++) K4[i] = 0.f;
  for (int r = 0; r < 3; r++)
    for (int c = 0; c < 3; c++) K4[r * 4 + c] = Ksm[f * 9 + r * 3 + c];
  K4[15] = 1.f;
  float M[16];
  for (int r = 0; r < 4; r++)
    for (int c = 0; c < 4; c++) {
      float acc = 0.f;
      for (int j = 0; j < 4; j++) acc += K4[r * 4 + j] * vm[f * 16 + j * 4 + c];
      M[r * 4 + c] = acc;
      P[f * 16 + r * 4 + c] = acc;
    }
  float a00=M[0],a01=M[1],a02=M[2],t0=M[3];
  float a10=M[4],a11=M[5],a12=M[6],t1=M[7];
  float a20=M[8],a21=M[9],a22=M[10],t2=M[11];
  float c00 =  a11*a22 - a12*a21;
  float c01 = -(a10*a22 - a12*a20);
  float c02 =  a10*a21 - a11*a20;
  float det = a00*c00 + a01*c01 + a02*c02;
  float id = 1.f / det;
  float i00 = c00*id, i01 = -(a01*a22 - a02*a21)*id, i02 =  (a01*a12 - a02*a11)*id;
  float i10 = c01*id, i11 =  (a00*a22 - a02*a20)*id, i12 = -(a00*a12 - a02*a10)*id;
  float i20 = c02*id, i21 = -(a00*a21 - a01*a20)*id, i22 =  (a00*a11 - a01*a10)*id;
  float ti0 = -(i00*t0 + i01*t1 + i02*t2);
  float ti1 = -(i10*t0 + i11*t1 + i12*t2);
  float ti2 = -(i20*t0 + i21*t1 + i22*t2);
  float* Pi = Pinv + f * 16;
  Pi[0]=i00; Pi[1]=i01; Pi[2]=i02;  Pi[3]=ti0;
  Pi[4]=i10; Pi[5]=i11; Pi[6]=i12;  Pi[7]=ti1;
  Pi[8]=i20; Pi[9]=i21; Pi[10]=i22; Pi[11]=ti2;
  Pi[12]=0;  Pi[13]=0;  Pi[14]=0;   Pi[15]=1;
}

// ---------------- RoPE + PRoPE, build attention operands ----------------
__global__ void rope_prope_kernel(
    const float* __restrict__ qf, const float* __restrict__ kf,
    const float* __restrict__ vf, const float* __restrict__ fcos,
    const float* __restrict__ fsin, const float* __restrict__ Pm,
    const float* __restrict__ Pinv, _Float16* __restrict__ Qall,
    _Float16* __restrict__ Kall, _Float16* __restrict__ Vall)
{
  const int s = blockIdx.x, h = blockIdx.y, d = threadIdx.x;
  const size_t base = (size_t)s * DIMC + h * DHC;
  const int pair = d & ~1, odd = d & 1;
  float c  = fcos[(size_t)s * DHC + pair];
  float sn = fsin[(size_t)s * DHC + pair + 1];
  float qe = qf[base + pair], qo = qf[base + pair + 1];
  float ke = kf[base + pair], ko = kf[base + pair + 1];
  float q_rope = odd ? (qe * sn + qo * c) : (qe * c - qo * sn);
  float k_rope = odd ? (ke * sn + ko * c) : (ke * c - ko * sn);
  float vv = vf[base + d];

  const int frame = s / TPFC;
  const float* P  = Pm + frame * 16;
  const float* Pi = Pinv + frame * 16;
  const int g = (d >> 2) * 4, j = d & 3;
  float q_pr = 0.f, k_pr = 0.f, v_pr = 0.f;
  #pragma unroll
  for (int i = 0; i < 4; i++) {
    q_pr += qf[base + g + i] * P[i * 4 + j];    // x @ P
    k_pr += kf[base + g + i] * Pi[j * 4 + i];   // x @ Pinv^T
    v_pr += vf[base + g + i] * Pi[j * 4 + i];
  }
  size_t qi0 = (((size_t)0 * HN + h) * SN + s) * DHC + d;
  size_t qi1 = (((size_t)1 * HN + h) * SN + s) * DHC + d;
  Qall[qi0] = (_Float16)q_rope;
  Qall[qi1] = (_Float16)q_pr;
  size_t ki0 = (((size_t)0 * HN + h) * SKVN + (SCC + s)) * DHC + d;
  size_t ki1 = (((size_t)1 * HN + h) * SKVN + (SCC + s)) * DHC + d;
  Kall[ki0] = (_Float16)k_rope;
  Kall[ki1] = (_Float16)k_pr;
  Vall[ki0] = (_Float16)vv;
  Vall[ki1] = (_Float16)v_pr;
}

__global__ void cache_split_kernel(const float* __restrict__ ck,
                                   const float* __restrict__ cv,
                                   _Float16* __restrict__ Kall,
                                   _Float16* __restrict__ Vall)
{
  const int s = blockIdx.x, h = blockIdx.y, d = threadIdx.x;
  size_t src = (((size_t)h * SCC) + s) * (2 * DHC);
  size_t d0 = (((size_t)0 * HN + h) * SKVN + s) * DHC + d;
  size_t d1 = (((size_t)1 * HN + h) * SKVN + s) * DHC + d;
  Kall[d0] = (_Float16)ck[src + d];
  Kall[d1] = (_Float16)ck[src + DHC + d];
  Vall[d0] = (_Float16)cv[src + d];
  Vall[d1] = (_Float16)cv[src + DHC + d];
}

__global__ void attn_post_kernel(const float* __restrict__ Oall,
                                 const float* __restrict__ Pm,
                                 _Float16* __restrict__ Ar,
                                 _Float16* __restrict__ Ap)
{
  const int s = blockIdx.x, h = blockIdx.y, d = threadIdx.x;
  size_t o0 = (((size_t)0 * HN + h) * SN + s) * DHC;
  size_t o1 = (((size_t)1 * HN + h) * SN + s) * DHC;
  Ar[(size_t)s * DIMC + h * DHC + d] = (_Float16)Oall[o0 + d];
  const int frame = s / TPFC;
  const float* P = Pm + frame * 16;
  const int g = (d >> 2) * 4, j = d & 3;
  float acc = 0.f;
  #pragma unroll
  for (int i = 0; i < 4; i++) acc += Oall[o1 + g + i] * P[j * 4 + i];  // x @ P^T
  Ap[(size_t)s * DIMC + h * DHC + d] = (_Float16)acc;
}

__global__ void split_heads_kernel(const float* __restrict__ src,
                                   _Float16* __restrict__ dst, int rows)
{
  const int s = blockIdx.x, h = blockIdx.y, d = threadIdx.x;
  dst[(((size_t)h * rows) + s) * DHC + d] =
      (_Float16)src[(size_t)s * DIMC + h * DHC + d];
}

__global__ void merge_heads_kernel(const float* __restrict__ src,
                                   _Float16* __restrict__ dst, int rows)
{
  const int s = blockIdx.x, h = blockIdx.y, d = threadIdx.x;
  dst[(size_t)s * DIMC + h * DHC + d] =
      (_Float16)src[(((size_t)h * rows) + s) * DHC + d];
}

__global__ void resgate_kernel(const float* __restrict__ base,
                               const float* __restrict__ t,
                               const float* __restrict__ sstT,
                               const float* __restrict__ temb,
                               int gateRow, float* __restrict__ out, size_t n)
{
  size_t idx = (size_t)blockIdx.x * 256 + threadIdx.x;
  if (idx >= n) return;
  int c = (int)(idx % DIMC);
  float g = sstT[gateRow * DIMC + c] + temb[gateRow * DIMC + c];
  out[idx] = base[idx] + t[idx] * g;
}

__global__ void cast_f16_kernel(const float* __restrict__ s,
                                _Float16* __restrict__ d, size_t n)
{
  size_t idx = (size_t)blockIdx.x * 256 + threadIdx.x;
  if (idx < n) d[idx] = (_Float16)s[idx];
}

// ---------------- host orchestration ----------------
extern "C" void kernel_launch(void* const* d_in, const int* in_sizes, int n_in,
                              void* d_out, int out_size, void* d_ws, size_t ws_size,
                              hipStream_t stream)
{
  (void)in_sizes; (void)n_in; (void)out_size; (void)ws_size;
  const float* hidden  = (const float*)d_in[0];
  const float* enc     = (const float*)d_in[1];
  const float* temb    = (const float*)d_in[2];
  const float* fcos    = (const float*)d_in[3];
  const float* fsin    = (const float*)d_in[4];
  const float* vmats   = (const float*)d_in[5];
  const float* Ksm     = (const float*)d_in[6];
  const float* cache_k = (const float*)d_in[7];
  const float* cache_v = (const float*)d_in[8];
  const float* sstT    = (const float*)d_in[9];
  const float* n2w     = (const float*)d_in[10];
  const float* n2b     = (const float*)d_in[11];
  const float* q1w = (const float*)d_in[12]; const float* q1b = (const float*)d_in[13];
  const float* k1w = (const float*)d_in[14]; const float* k1b = (const float*)d_in[15];
  const float* v1w = (const float*)d_in[16]; const float* v1b = (const float*)d_in[17];
  const float* o1w = (const float*)d_in[18]; const float* o1b = (const float*)d_in[19];
  const float* opw = (const float*)d_in[20]; const float* opb = (const float*)d_in[21];
  const float* q2w = (const float*)d_in[22]; const float* q2b = (const float*)d_in[23];
  const float* k2w = (const float*)d_in[24]; const float* k2b = (const float*)d_in[25];
  const float* v2w = (const float*)d_in[26]; const float* v2b = (const float*)d_in[27];
  const float* o2w = (const float*)d_in[28]; const float* o2b = (const float*)d_in[29];
  const float* nq1 = (const float*)d_in[30]; const float* nk1 = (const float*)d_in[31];
  const float* nq2 = (const float*)d_in[32]; const float* nk2 = (const float*)d_in[33];
  const float* ff1w = (const float*)d_in[34]; const float* ff1b = (const float*)d_in[35];
  const float* ff2w = (const float*)d_in[36]; const float* ff2b = (const float*)d_in[37];
  float* out = (float*)d_out;

  char* base = (char*)d_ws; size_t off = 0;
  auto alloc = [&](size_t bytes) -> char* {
    char* p = base + off;
    off = (off + bytes + 255) & ~(size_t)255;
    return p;
  };
  _Float16* xn16  = (_Float16*)alloc((size_t)SN * DIMC * 2);   // xn / xn2 / xn3
  _Float16* enc16 = (_Float16*)alloc((size_t)LEN * DIMC * 2);
  float* fA   = (float*)alloc((size_t)SN * DIMC * 4);          // q / q2 / ff2-out
  float* fB   = (float*)alloc((size_t)SN * DIMC * 4);          // k / k2
  float* fC   = (float*)alloc((size_t)SN * DIMC * 4);          // v / v2
  float* fD   = (float*)alloc((size_t)SN * DIMC * 4);          // o1+op accumulator
  float* hbuf = (float*)alloc((size_t)SN * DIMC * 4);          // residual stream
  _Float16* Qall = (_Float16*)alloc((size_t)2 * HN * SN * DHC * 2);   // also Q2
  _Float16* Kall = (_Float16*)alloc((size_t)2 * HN * SKVN * DHC * 2); // also K2
  _Float16* Vall = (_Float16*)alloc((size_t)2 * HN * SKVN * DHC * 2); // also V2
  float*    Oall = (float*)alloc((size_t)2 * HN * SN * DHC * 4);      // also O2
  _Float16* Ar16 = (_Float16*)alloc((size_t)SN * DIMC * 2);           // also o2 merged
  _Float16* Ap16 = (_Float16*)alloc((size_t)SN * DIMC * 2);
  _Float16* ff16 = (_Float16*)alloc((size_t)SN * FFNN * 2);
  float* Pm   = (float*)alloc(FNUM * 16 * 4);
  float* Pinv = (float*)alloc(FNUM * 16 * 4);

  auto gemmF32 = [&](const _Float16* A, const float* W, const float* bias,
                     float* o, int M, int N, int K) {
    dim3 g((M + 127) / 128, N / 64);
    gemm_kernel<false, 0><<<g, 256, 0, stream>>>(A, W, bias, o, M, N, K);
  };
  auto gemmAcc = [&](const _Float16* A, const float* W, const float* bias,
                     float* o, int M, int N, int K) {
    dim3 g((M + 127) / 128, N / 64);
    gemm_kernel<false, 2><<<g, 256, 0, stream>>>(A, W, bias, o, M, N, K);
  };
  auto gemmGelu16 = [&](const _Float16* A, const float* W, const float* bias,
                        _Float16* o, int M, int N, int K) {
    dim3 g((M + 127) / 128, N / 64);
    gemm_kernel<true, 1><<<g, 256, 0, stream>>>(A, W, bias, o, M, N, K);
  };

  // ---- phase 0: P/P^-1 and AdaLN-modulated LN1 ----
  prope_mats_kernel<<<1, 32, 0, stream>>>(vmats, Ksm, Pm, Pinv);
  layernorm_kernel<<<SN, 256, 0, stream>>>(hidden, xn16, nullptr, nullptr,
                                           sstT, temb, 1, 0);
  // ---- phase 1: self-attn projections + RMS ----
  gemmF32(xn16, q1w, q1b, fA, SN, DIMC, DIMC);
  gemmF32(xn16, k1w, k1b, fB, SN, DIMC, DIMC);
  gemmF32(xn16, v1w, v1b, fC, SN, DIMC, DIMC);
  rms_kernel<<<SN, 256, 0, stream>>>(fA, nq1);
  rms_kernel<<<SN, 256, 0, stream>>>(fB, nk1);
  // ---- phase 2: attention operands (cache + rope + prope) ----
  cache_split_kernel<<<dim3(SCC, HN), 128, 0, stream>>>(cache_k, cache_v, Kall, Vall);
  rope_prope_kernel<<<dim3(SN, HN), 128, 0, stream>>>(fA, fB, fC, fcos, fsin,
                                                      Pm, Pinv, Qall, Kall, Vall);
  // ---- phase 3: dual-stream flash attention ----
  flash_kernel<<<dim3(2 * HN, (SN + 63) / 64), 128, 0, stream>>>(
      Qall, Kall, Vall, Oall, SN, SKVN, ATT_SCALE);
  attn_post_kernel<<<dim3(SN, HN), 128, 0, stream>>>(Oall, Pm, Ar16, Ap16);
  // ---- phase 4: output projections + gated residual ----
  gemmF32(Ar16, o1w, o1b, fD, SN, DIMC, DIMC);
  gemmAcc(Ap16, opw, opb, fD, SN, DIMC, DIMC);
  {
    size_t n = (size_t)SN * DIMC;
    resgate_kernel<<<(unsigned)((n + 255) / 256), 256, 0, stream>>>(
        hidden, fD, sstT, temb, 2, hbuf, n);
  }
  // ---- phase 5: cross-attention ----
  layernorm_kernel<<<SN, 256, 0, stream>>>(hbuf, xn16, n2w, n2b,
                                           nullptr, nullptr, 0, 0);
  gemmF32(xn16, q2w, q2b, fA, SN, DIMC, DIMC);
  rms_kernel<<<SN, 256, 0, stream>>>(fA, nq2);
  {
    size_t n = (size_t)LEN * DIMC;
    cast_f16_kernel<<<(unsigned)((n + 255) / 256), 256, 0, stream>>>(enc, enc16, n);
  }
  gemmF32(enc16, k2w, k2b, fB, LEN, DIMC, DIMC);
  rms_kernel<<<LEN, 256, 0, stream>>>(fB, nk2);
  gemmF32(enc16, v2w, v2b, fC, LEN, DIMC, DIMC);
  split_heads_kernel<<<dim3(SN, HN), 128, 0, stream>>>(fA, Qall, SN);
  split_heads_kernel<<<dim3(LEN, HN), 128, 0, stream>>>(fB, Kall, LEN);
  split_heads_kernel<<<dim3(LEN, HN), 128, 0, stream>>>(fC, Vall, LEN);
  flash_kernel<<<dim3(HN, (SN + 63) / 64), 128, 0, stream>>>(
      Qall, Kall, Vall, Oall, SN, LEN, ATT_SCALE);
  merge_heads_kernel<<<dim3(SN, HN), 128, 0, stream>>>(Oall, Ar16, SN);
  gemmAcc(Ar16, o2w, o2b, hbuf, SN, DIMC, DIMC);   // h += o2 proj, fused
  // ---- phase 6: FFN ----
  layernorm_kernel<<<SN, 256, 0, stream>>>(hbuf, xn16, nullptr, nullptr,
                                           sstT, temb, 4, 3);
  gemmGelu16(xn16, ff1w, ff1b, ff16, SN, FFNN, DIMC);
  gemmF32(ff16, ff2w, ff2b, fA, SN, DIMC, FFNN);
  {
    size_t n = (size_t)SN * DIMC;
    resgate_kernel<<<(unsigned)((n + 255) / 256), 256, 0, stream>>>(
        hbuf, fA, sstT, temb, 5, out, n);
  }
}
